// SanaBlock_16552803958857
// MI455X (gfx1250) — matrix operand order repaired and checked by reading, not compiled, not run on hardware
//
#include <hip/hip_runtime.h>
#include <hip/hip_bf16.h>
#include <cstdint>

// ---------------------------------------------------------------------------
// SanaBlock for MI455X (gfx1250, wave32, WMMA + Tensor Data Mover).
// GEMMs: TDM double-buffered LDS staging -> ds_load fragments ->
// v_wmma_f32_16x16x32_f16, fused epilogues. Glue ops are VALU kernels.
// ---------------------------------------------------------------------------

#define C_DIM   1152
#define N_TOK   4096
#define B_DIM   2
#define M_CTX   300
#define HID     2880        // GLUMBConv hidden
#define H2      5760        // 2*HID
#define HEADS_LA 36
#define D_LA     32
#define HEADS_X  16
#define D_X      72
#define GSTRIDE  (6 * C_DIM)   // per-batch stride inside modulation buffer

// GEMM block tiling
#define BM 128
#define BN 64
#define BK 32

typedef __attribute__((ext_vector_type(16))) _Float16 v16h;
typedef __attribute__((ext_vector_type(8)))  _Float16 h8;
typedef __attribute__((ext_vector_type(8)))  float    v8f;
typedef __attribute__((ext_vector_type(4)))  unsigned int u32x4;
typedef __attribute__((ext_vector_type(4)))  int      i32x4;
typedef __attribute__((ext_vector_type(8)))  int      i32x8;

union FragU { v16h v; h8 h[2]; };

// ---- WMMA fragment loaders (layouts per CDNA5 ISA §7.12.2) ------
// A 16x32 f16: lane L holds row M=L&15; halves 0..7 -> K = kb+0..7,
// halves 8..15 -> K = kb+16..23, where kb = (L&16)?8:0.
__device__ inline v16h load_a_frag(const _Float16* __restrict__ rowp, int lane) {
  const _Float16* p = rowp + ((lane & 16) ? 8 : 0);
  FragU f;
  f.h[0] = *(const h8*)(p);
  f.h[1] = *(const h8*)(p + 16);
  return f.v;
}
// B 32x16 f16: lane L holds column N=L&15; halves 0..7 -> K = kb+0..7,
// halves 8..15 -> K = kb+16..23, where kb = (L&16)?8:0.
__device__ inline v16h load_b_frag(const _Float16* __restrict__ colp, int lane) {
  const _Float16* p = colp + ((lane & 16) ? 8 : 0);
  FragU f;
  f.h[0] = *(const h8*)(p);
  f.h[1] = *(const h8*)(p + 16);
  return f.v;
}

// ---- Tensor Data Mover: 2-D f16 tile (tile_rows x tile_cols) -> LDS --------
// Descriptor layout per CDNA5 ISA (D# groups 0..3).
// tensor_rows clamps dim1 so OOB rows read as zero (edge M tiles).
// This toolchain's builtin takes 6 args: (g0, g1, g2, g3, g4, cpol).
__device__ inline void tdm_load_2d_f16(const _Float16* gptr, unsigned lds_off,
                                       int tile_cols, int tile_rows,
                                       int tensor_rows, long long row_stride)
{
  unsigned long long ga = (unsigned long long)(uintptr_t)gptr;
  unsigned dim0 = (unsigned)tile_cols;
  unsigned dim1 = (unsigned)(tensor_rows < 0 ? 0 : tensor_rows);
  unsigned long long s0 = (unsigned long long)row_stride;

  u32x4 g0;
  g0[0] = 1u;                                            // count=1, user desc
  g0[1] = lds_off;                                       // lds_addr (bytes)
  g0[2] = (unsigned)ga;                                  // global_addr[31:0]
  g0[3] = ((unsigned)(ga >> 32) & 0x1FFFFFFu) | (2u << 30); // [56:32] | type=2

  i32x8 g1;
  g1[0] = (int)(1u << 16);                               // data_size=1 (2 B)
  g1[1] = (int)((dim0 & 0xFFFFu) << 16);                 // tensor_dim0[15:0]
  g1[2] = (int)(((dim0 >> 16) & 0xFFFFu) | ((dim1 & 0xFFFFu) << 16));
  g1[3] = (int)(((dim1 >> 16) & 0xFFFFu) | ((dim0 & 0xFFFFu) << 16)); // tile_dim0
  g1[4] = (int)((unsigned)tile_rows & 0xFFFFu);          // tile_dim1
  g1[5] = (int)(unsigned)(s0 & 0xFFFFFFFFull);           // dim0_stride[31:0]
  g1[6] = (int)(unsigned)((s0 >> 32) & 0xFFFFull);       // dim0_stride[47:32]
  g1[7] = 0;

  i32x4 gz4 = {0, 0, 0, 0};                              // groups 2/3: 2-D tensor
  i32x8 gz8 = {0, 0, 0, 0, 0, 0, 0, 0};                  // extra group (unused)
  __builtin_amdgcn_tensor_load_to_lds(g0, g1, gz4, gz4, gz8, 0);
}

// ---- GEMM: D[M,N] = A[M,K] @ W[N,K]^T, TDM-staged LDS, fused epilogues -----
// MODE 0: out f16 = acc
// MODE 1: out f16 = acc + bias
// MODE 2: out f16 = silu(acc + bias)
// MODE 3: out f32 = residual + gate * (acc + bias)   (bias/gate optional)
template <int MODE>
__global__ __launch_bounds__(256) void gemm_wmma_kernel(
    const _Float16* __restrict__ A, const _Float16* __restrict__ W,
    const float* __restrict__ bias, const float* __restrict__ gate,
    const float* __restrict__ residual,
    _Float16* __restrict__ outh, float* __restrict__ outf,
    int M, int Ncols, int K, int rows_per_batch)
{
  __shared__ _Float16 As[2][BM * BK];   // 8 KB per stage
  __shared__ _Float16 Bs[2][BN * BK];   // 4 KB per stage

  const int lane = threadIdx.x & 31;
  const int wave = __builtin_amdgcn_readfirstlane((int)(threadIdx.x >> 5)); // 0..7
  const int wm = wave >> 2;             // 2 wave rows  (64 M-rows each)
  const int wn = wave & 3;              // 4 wave cols  (16 N-cols each)
  const int tileMB = blockIdx.y * BM;
  const int tileNB = blockIdx.x * BN;

  const int aRow = wave * 16;           // this wave DMAs 16 rows of A tile
  const int bRow = wave * 8;            // ... and 8 rows of B tile

  const _Float16* __restrict__ aG = A + (size_t)(tileMB + aRow) * K;
  const _Float16* __restrict__ bG = W + (size_t)(tileNB + bRow) * K;
  const int aRemain = M - (tileMB + aRow);   // rows still in-bounds (may be <=0)

  v8f acc[4] = {};

  // prologue: stage 0
  tdm_load_2d_f16(aG, (unsigned)(uintptr_t)&As[0][aRow * BK], BK, 16, aRemain, K);
  tdm_load_2d_f16(bG, (unsigned)(uintptr_t)&Bs[0][bRow * BK], BK, 8, 8, K);
  __builtin_amdgcn_s_wait_tensorcnt(0);
  __syncthreads();

  const int nStages = K / BK;
  for (int s = 0; s < nStages; ++s) {
    const int buf = s & 1;
    if (s + 1 < nStages) {               // kick next stage DMA (overlaps math)
      const int k1 = (s + 1) * BK;
      tdm_load_2d_f16(aG + k1, (unsigned)(uintptr_t)&As[buf ^ 1][aRow * BK],
                      BK, 16, aRemain, K);
      tdm_load_2d_f16(bG + k1, (unsigned)(uintptr_t)&Bs[buf ^ 1][bRow * BK],
                      BK, 8, 8, K);
    }
    // one B fragment reused across 4 A sub-tiles
    v16h bf = load_b_frag(&Bs[buf][(wn * 16 + (lane & 15)) * BK], lane);
#pragma unroll
    for (int t = 0; t < 4; ++t) {
      v16h af = load_a_frag(&As[buf][(wm * 64 + t * 16 + (lane & 15)) * BK], lane);
      acc[t] = __builtin_amdgcn_wmma_f32_16x16x32_f16(
          false, af, false, bf, (short)0, acc[t], false, false);
    }
    __builtin_amdgcn_s_wait_tensorcnt(0);
    __syncthreads();
  }

  // epilogue
  const int col = tileNB + wn * 16 + (lane & 15);
#pragma unroll
  for (int t = 0; t < 4; ++t) {
    const int rbase = tileMB + wm * 64 + t * 16 + ((lane & 16) ? 8 : 0);
#pragma unroll
    for (int r = 0; r < 8; ++r) {
      const int row = rbase + r;
      if (row >= M) continue;
      const size_t idx = (size_t)row * Ncols + col;
      float v = acc[t][r];
      if (MODE == 0) {
        outh[idx] = (_Float16)v;
      } else if (MODE == 1) {
        outh[idx] = (_Float16)(v + bias[col]);
      } else if (MODE == 2) {
        float u = v + bias[col];
        outh[idx] = (_Float16)(u / (1.f + __expf(-u)));   // silu
      } else {
        float u = v + (bias ? bias[col] : 0.f);
        float g = gate ? gate[(size_t)(row / rows_per_batch) * GSTRIDE + col] : 1.f;
        outf[idx] = residual[idx] + g * u;
      }
    }
  }
}

// ---- modulation vectors: m = sst + t  (B,6,C) ------------------------------
__global__ void addmod_kernel(const float* __restrict__ sst,
                              const float* __restrict__ t,
                              float* __restrict__ m, int total)
{
  int i = blockIdx.x * 256 + threadIdx.x;
  if (i < total) m[i] = sst[i % (6 * C_DIM)] + t[i];
}

// ---- generic f32 -> f16 convert --------------------------------------------
__global__ void f32_to_f16_kernel(const float* __restrict__ in,
                                  _Float16* __restrict__ out, int n)
{
  int i = blockIdx.x * 256 + threadIdx.x;
  if (i < n) out[i] = (_Float16)in[i];
}

// ---- LayerNorm (no affine) + adaLN scale/shift, f32 -> f16 -----------------
__global__ __launch_bounds__(256) void ln_mod_kernel(
    const float* __restrict__ x, const float* __restrict__ m,
    int sh_row, int sc_row, _Float16* __restrict__ out)
{
  const int row = blockIdx.x;          // b*N_TOK + n
  const int b = row >> 12;
  const float* __restrict__ xr = x + (size_t)row * C_DIM;
  float s = 0.f, ss = 0.f;
  for (int c = threadIdx.x; c < C_DIM; c += 256) {
    float v = xr[c]; s += v; ss += v * v;
  }
  __shared__ float r1[256], r2[256];
  r1[threadIdx.x] = s; r2[threadIdx.x] = ss;
  __syncthreads();
  for (int o = 128; o > 0; o >>= 1) {
    if (threadIdx.x < o) {
      r1[threadIdx.x] += r1[threadIdx.x + o];
      r2[threadIdx.x] += r2[threadIdx.x + o];
    }
    __syncthreads();
  }
  const float mu  = r1[0] * (1.f / C_DIM);
  const float var = r2[0] * (1.f / C_DIM) - mu * mu;
  const float rs  = rsqrtf(var + 1e-6f);
  const float* __restrict__ shv = m + (size_t)b * GSTRIDE + (size_t)sh_row * C_DIM;
  const float* __restrict__ scv = m + (size_t)b * GSTRIDE + (size_t)sc_row * C_DIM;
  _Float16* __restrict__ op = out + (size_t)row * C_DIM;
  for (int c = threadIdx.x; c < C_DIM; c += 256)
    op[c] = (_Float16)((xr[c] - mu) * rs * (1.f + scv[c]) + shv[c]);
}

// ---- LiteLA: vk[b,h,e,dd] = sum_n Vpad[e,n] * relu(K[dd,n]) ---------------
__global__ __launch_bounds__(128) void litela_vk_kernel(
    const _Float16* __restrict__ qkv, float* __restrict__ vk)
{
  const int id = blockIdx.x;                    // (b*36 + h)*33 + e
  const int e = id % 33;
  const int h = (id / 33) % HEADS_LA;
  const int b = id / (33 * HEADS_LA);
  const int lane = threadIdx.x & 31;
  const int wave = threadIdx.x >> 5;

  float acc[32];
#pragma unroll
  for (int j = 0; j < 32; ++j) acc[j] = 0.f;

  for (int n = threadIdx.x; n < N_TOK; n += 128) {
    const _Float16* __restrict__ rowp = qkv + ((size_t)(b * N_TOK + n)) * (3 * C_DIM);
    float vv = (e < D_LA) ? (float)rowp[2 * C_DIM + h * D_LA + e] : 1.f;
    const _Float16* __restrict__ kp = rowp + C_DIM + h * D_LA;
#pragma unroll
    for (int j = 0; j < 32; ++j) {
      float kk = (float)kp[j];
      kk = kk > 0.f ? kk : 0.f;
      acc[j] += vv * kk;
    }
  }
  __shared__ float red[4][32];
#pragma unroll
  for (int j = 0; j < 32; ++j) {
    float s = acc[j];
    for (int o = 16; o > 0; o >>= 1) s += __shfl_down(s, o, 32);
    if (lane == 0) red[wave][j] = s;
  }
  __syncthreads();
  if (threadIdx.x < 32) {
    float s = red[0][threadIdx.x] + red[1][threadIdx.x] +
              red[2][threadIdx.x] + red[3][threadIdx.x];
    vk[(size_t)id * 32 + threadIdx.x] = s;
  }
}

// ---- LiteLA: out[e,n] = (vk @ relu(q))[e] / (den + eps) --------------------
__global__ __launch_bounds__(256) void litela_out_kernel(
    const _Float16* __restrict__ qkv, const float* __restrict__ vk,
    _Float16* __restrict__ attn)
{
  const int row = blockIdx.x;          // b*N_TOK + n
  const int b = row >> 12;
  __shared__ float qsh[C_DIM];
  __shared__ float den[HEADS_LA];
  const _Float16* __restrict__ qr = qkv + (size_t)row * (3 * C_DIM);
  for (int c = threadIdx.x; c < C_DIM; c += 256) {
    float q = (float)qr[c];
    qsh[c] = q > 0.f ? q : 0.f;
  }
  __syncthreads();
  if (threadIdx.x < HEADS_LA) {
    const int h = threadIdx.x;
    const float* __restrict__ vkp = vk + (((size_t)(b * HEADS_LA + h)) * 33 + 32) * 32;
    float s = 0.f;
#pragma unroll
    for (int dd = 0; dd < 32; ++dd) s += vkp[dd] * qsh[h * D_LA + dd];
    den[h] = s + 1e-8f;
  }
  __syncthreads();
  for (int idx = threadIdx.x; idx < C_DIM; idx += 256) {
    const int h = idx >> 5, e = idx & 31;
    const float* __restrict__ vkp = vk + (((size_t)(b * HEADS_LA + h)) * 33 + e) * 32;
    float s = 0.f;
#pragma unroll
    for (int dd = 0; dd < 32; ++dd) s += vkp[dd] * qsh[h * D_LA + dd];
    attn[(size_t)row * C_DIM + idx] = (_Float16)(s / den[h]);
  }
}

// ---- Cross-attention: 16 heads, 300 keys, softmax per (b,n,h) --------------
__global__ __launch_bounds__(256) void cross_attn_kernel(
    const _Float16* __restrict__ q, const _Float16* __restrict__ kv,
    _Float16* __restrict__ out)
{
  const int row = blockIdx.x;          // b*N_TOK + n
  const int b = row >> 12;
  const int tid = threadIdx.x;
  __shared__ float qs[C_DIM];
  __shared__ float sc[M_CTX];
  __shared__ float red[256];
  const _Float16* __restrict__ qr = q + (size_t)row * C_DIM;
  for (int c = tid; c < C_DIM; c += 256) qs[c] = (float)qr[c];
  __syncthreads();
  const float scale = 0.117851130f;    // 1/sqrt(72)

  for (int h = 0; h < HEADS_X; ++h) {
    const int qo = h * D_X;
    for (int m = tid; m < M_CTX; m += 256) {
      const _Float16* __restrict__ kr = kv + (size_t)(b * M_CTX + m) * (2 * C_DIM) + qo;
      float s = 0.f;
#pragma unroll
      for (int j = 0; j < D_X; ++j) s += qs[qo + j] * (float)kr[j];
      sc[m] = s * scale;
    }
    __syncthreads();
    float lm = -1e30f;
    for (int m = tid; m < M_CTX; m += 256) lm = fmaxf(lm, sc[m]);
    red[tid] = lm; __syncthreads();
    for (int o = 128; o > 0; o >>= 1) {
      if (tid < o) red[tid] = fmaxf(red[tid], red[tid + o]);
      __syncthreads();
    }
    const float mx = red[0];
    __syncthreads();
    float ls = 0.f;
    for (int m = tid; m < M_CTX; m += 256) {
      float e = __expf(sc[m] - mx);
      sc[m] = e; ls += e;
    }
    red[tid] = ls; __syncthreads();
    for (int o = 128; o > 0; o >>= 1) {
      if (tid < o) red[tid] += red[tid + o];
      __syncthreads();
    }
    const float inv = 1.f / red[0];
    __syncthreads();
    for (int c = tid; c < D_X; c += 256) {
      float o = 0.f;
      for (int m = 0; m < M_CTX; ++m)
        o += sc[m] * (float)kv[(size_t)(b * M_CTX + m) * (2 * C_DIM) + C_DIM + qo + c];
      out[(size_t)row * C_DIM + qo + c] = (_Float16)(o * inv);
    }
    __syncthreads();
  }
}

// ---- Depthwise 3x3 (SAME) + bias + GLU: glu = a * silu(g) ------------------
__global__ __launch_bounds__(256) void dwglu_kernel(
    const _Float16* __restrict__ hbuf, const float* __restrict__ dw_w,
    const float* __restrict__ dw_b, _Float16* __restrict__ glu)
{
  const int row = blockIdx.x;          // b*N_TOK + n
  const int b = row >> 12;
  const int n = row & (N_TOK - 1);
  const int yp = n >> 6, xp = n & 63;
  for (int ch = threadIdx.x; ch < HID; ch += 256) {
    float a = dw_b[ch];
    float g = dw_b[ch + HID];
#pragma unroll
    for (int dy = -1; dy <= 1; ++dy) {
      const int yy = yp + dy;
      if (yy < 0 || yy > 63) continue;
#pragma unroll
      for (int dx = -1; dx <= 1; ++dx) {
        const int xx = xp + dx;
        if (xx < 0 || xx > 63) continue;
        const _Float16* __restrict__ ip =
            hbuf + ((size_t)(b * N_TOK + (yy << 6) + xx)) * H2;
        const int widx = (dy + 1) * 3 + (dx + 1);
        a += dw_w[(size_t)ch * 9 + widx]        * (float)ip[ch];
        g += dw_w[(size_t)(ch + HID) * 9 + widx] * (float)ip[ch + HID];
      }
    }
    const float sg = g / (1.f + __expf(-g));   // silu(g)
    glu[(size_t)row * HID + ch] = (_Float16)(a * sg);
  }
}

// ---------------------------------------------------------------------------
extern "C" void kernel_launch(void* const* d_in, const int* in_sizes, int n_in,
                              void* d_out, int out_size, void* d_ws, size_t ws_size,
                              hipStream_t stream) {
  (void)in_sizes; (void)n_in; (void)out_size; (void)ws_size;
  const float* x       = (const float*)d_in[0];
  const float* y       = (const float*)d_in[1];
  const float* t       = (const float*)d_in[2];
  const float* sst     = (const float*)d_in[3];
  const float* qkv_w   = (const float*)d_in[4];
  const float* aproj_w = (const float*)d_in[5];
  const float* aproj_b = (const float*)d_in[6];
  const float* q_w     = (const float*)d_in[7];
  const float* q_b     = (const float*)d_in[8];
  const float* kv_w    = (const float*)d_in[9];
  const float* kv_b    = (const float*)d_in[10];
  const float* cproj_w = (const float*)d_in[11];
  const float* cproj_b = (const float*)d_in[12];
  const float* inv_w   = (const float*)d_in[13];
  const float* inv_b   = (const float*)d_in[14];
  const float* dw_w    = (const float*)d_in[15];
  const float* dw_b    = (const float*)d_in[16];
  const float* pw_w    = (const float*)d_in[17];
  float* out = (float*)d_out;

  const int ROWS = B_DIM * N_TOK;       // 8192
  const int YROWS = B_DIM * M_CTX;      // 600

  // ---- workspace carve-out (256B aligned) ----
  char* p = (char*)d_ws;
  auto alloc_h = [&](size_t n) { _Float16* r = (_Float16*)p; p += ((n * 2 + 255) & ~(size_t)255); return r; };
  auto alloc_f = [&](size_t n) { float*     r = (float*)p;     p += ((n * 4 + 255) & ~(size_t)255); return r; };

  _Float16* w_qkv   = alloc_h((size_t)3 * C_DIM * C_DIM);
  _Float16* w_aproj = alloc_h((size_t)C_DIM * C_DIM);
  _Float16* w_q     = alloc_h((size_t)C_DIM * C_DIM);
  _Float16* w_kv    = alloc_h((size_t)2 * C_DIM * C_DIM);
  _Float16* w_cproj = alloc_h((size_t)C_DIM * C_DIM);
  _Float16* w_inv   = alloc_h((size_t)H2 * C_DIM);
  _Float16* w_pw    = alloc_h((size_t)C_DIM * HID);

  _Float16* xa_h    = alloc_h((size_t)ROWS * C_DIM);
  _Float16* qkv_h   = alloc_h((size_t)ROWS * 3 * C_DIM);
  _Float16* attn_h  = alloc_h((size_t)ROWS * C_DIM);
  _Float16* x1_h    = alloc_h((size_t)ROWS * C_DIM);
  _Float16* qx_h    = alloc_h((size_t)ROWS * C_DIM);
  _Float16* y_h     = alloc_h((size_t)YROWS * C_DIM);
  _Float16* kv_h    = alloc_h((size_t)YROWS * 2 * C_DIM);
  _Float16* xatt_h  = alloc_h((size_t)ROWS * C_DIM);
  _Float16* xm_h    = alloc_h((size_t)ROWS * C_DIM);
  _Float16* hinv_h  = alloc_h((size_t)ROWS * H2);
  _Float16* glu_h   = alloc_h((size_t)ROWS * HID);

  float* m_buf = alloc_f((size_t)B_DIM * 6 * C_DIM);
  float* vk_buf = alloc_f((size_t)B_DIM * HEADS_LA * 33 * 32);
  float* x1_f  = alloc_f((size_t)ROWS * C_DIM);

  auto cdiv = [](int a, int b) { return (a + b - 1) / b; };
  auto cvt = [&](const float* src, _Float16* dst, int n) {
    f32_to_f16_kernel<<<cdiv(n, 256), 256, 0, stream>>>(src, dst, n);
  };

  // 0) weight conversion to f16
  cvt(qkv_w,   w_qkv,   3 * C_DIM * C_DIM);
  cvt(aproj_w, w_aproj, C_DIM * C_DIM);
  cvt(q_w,     w_q,     C_DIM * C_DIM);
  cvt(kv_w,    w_kv,    2 * C_DIM * C_DIM);
  cvt(cproj_w, w_cproj, C_DIM * C_DIM);
  cvt(inv_w,   w_inv,   H2 * C_DIM);
  cvt(pw_w,    w_pw,    C_DIM * HID);

  // 1) modulation vectors m = sst + t
  addmod_kernel<<<cdiv(B_DIM * 6 * C_DIM, 256), 256, 0, stream>>>(
      sst, t, m_buf, B_DIM * 6 * C_DIM);

  // 2) xa = ln(x)*(1+sc_a) + sh_a  (rows 0=sh_a, 1=sc_a)
  ln_mod_kernel<<<ROWS, 256, 0, stream>>>(x, m_buf, 0, 1, xa_h);

  // 3) qkv = xa @ qkv_w^T      (WMMA + TDM)
  gemm_wmma_kernel<0><<<dim3(3 * C_DIM / BN, ROWS / BM), 256, 0, stream>>>(
      xa_h, w_qkv, nullptr, nullptr, nullptr, qkv_h, nullptr,
      ROWS, 3 * C_DIM, C_DIM, N_TOK);

  // 4) LiteLA: vk reduction then normalized output
  litela_vk_kernel<<<B_DIM * HEADS_LA * 33, 128, 0, stream>>>(qkv_h, vk_buf);
  litela_out_kernel<<<ROWS, 256, 0, stream>>>(qkv_h, vk_buf, attn_h);

  // 5) x1 = x + g_a * (attn @ aproj_w^T + aproj_b)
  gemm_wmma_kernel<3><<<dim3(C_DIM / BN, ROWS / BM), 256, 0, stream>>>(
      attn_h, w_aproj, aproj_b, m_buf + 2 * C_DIM, x, nullptr, x1_f,
      ROWS, C_DIM, C_DIM, N_TOK);

  // 6) cross attention q/kv projections
  cvt(x1_f, x1_h, ROWS * C_DIM);
  gemm_wmma_kernel<1><<<dim3(C_DIM / BN, ROWS / BM), 256, 0, stream>>>(
      x1_h, w_q, q_b, nullptr, nullptr, qx_h, nullptr,
      ROWS, C_DIM, C_DIM, N_TOK);
  cvt(y, y_h, YROWS * C_DIM);
  gemm_wmma_kernel<1><<<dim3(2 * C_DIM / BN, cdiv(YROWS, BM)), 256, 0, stream>>>(
      y_h, w_kv, kv_b, nullptr, nullptr, kv_h, nullptr,
      YROWS, 2 * C_DIM, C_DIM, M_CTX);

  // 7) softmax cross attention
  cross_attn_kernel<<<ROWS, 256, 0, stream>>>(qx_h, kv_h, xatt_h);

  // 8) x2 = x1 + (xatt @ cproj_w^T + cproj_b)  -> d_out holds x2
  gemm_wmma_kernel<3><<<dim3(C_DIM / BN, ROWS / BM), 256, 0, stream>>>(
      xatt_h, w_cproj, cproj_b, nullptr, x1_f, nullptr, out,
      ROWS, C_DIM, C_DIM, N_TOK);

  // 9) GLUMBConv: xm = ln(x2)*(1+sc_m)+sh_m  (rows 3=sh_m, 4=sc_m)
  ln_mod_kernel<<<ROWS, 256, 0, stream>>>(out, m_buf, 3, 4, xm_h);

  // 10) hinv = silu(xm @ inv_w^T + inv_b)
  gemm_wmma_kernel<2><<<dim3(H2 / BN, ROWS / BM), 256, 0, stream>>>(
      xm_h, w_inv, inv_b, nullptr, nullptr, hinv_h, nullptr,
      ROWS, H2, C_DIM, N_TOK);

  // 11) depthwise 3x3 + bias + GLU
  dwglu_kernel<<<ROWS, 256, 0, stream>>>(hinv_h, dw_w, dw_b, glu_h);

  // 12) out = x2 + g_m * (glu @ pw_w^T)   (residual aliases d_out: each thread
  //     reads its own element before writing it)
  gemm_wmma_kernel<3><<<dim3(C_DIM / BN, ROWS / BM), 256, 0, stream>>>(
      glu_h, w_pw, nullptr, m_buf + 5 * C_DIM, out, nullptr, out,
      ROWS, C_DIM, HID, N_TOK);
}
